// MinGRUBlock_22187801051583
// MI455X (gfx1250) — matrix operand order repaired and checked by reading, not compiled, not run on hardware
//
#include <hip/hip_runtime.h>

#define B_  4
#define T_  8192
#define H_  512
#define NROW (B_ * T_)          // 32768 rows
#define NCHUNK 16
#define CHUNK  (T_ / NCHUNK)    // 512 steps per chunk
#define BH (B_ * H_)            // 2048 scan channels
#define MBLK 4                  // 16x16 M-tiles per wave (64-row strip)

typedef __attribute__((ext_vector_type(16))) __bf16 v16bf;
typedef __attribute__((ext_vector_type(8)))  __bf16 v8bf;
typedef __attribute__((ext_vector_type(8)))  float  v8f;

__device__ __forceinline__ float softplus_f(float x) {
    // stable: max(x,0) + log1p(exp(-|x|))
    return fmaxf(x, 0.0f) + log1pf(__expf(-fabsf(x)));
}
__device__ __forceinline__ float logaddexp_f(float a, float b) {
    float m = fmaxf(a, b);
    return m + log1pf(__expf(fminf(a, b) - m));
}

// ---------------------------------------------------------------------------
// LayerNorm (fp32 in) -> bf16 out, one block per row of H=512 (256 thr x 2)
// ---------------------------------------------------------------------------
__global__ __launch_bounds__(256)
void ln_to_bf16_kernel(const float* __restrict__ x,
                       const float* __restrict__ g,
                       const float* __restrict__ bta,
                       __bf16* __restrict__ out) {
    __shared__ float red[256];
    const int row = blockIdx.x;
    const int tid = threadIdx.x;
    const float* xr = x + (size_t)row * H_;
    float2 v = *(const float2*)(xr + tid * 2);

    red[tid] = v.x + v.y;
    __syncthreads();
    #pragma unroll
    for (int off = 128; off > 0; off >>= 1) {
        if (tid < off) red[tid] += red[tid + off];
        __syncthreads();
    }
    const float mu = red[0] * (1.0f / H_);
    __syncthreads();

    const float d0 = v.x - mu, d1 = v.y - mu;
    red[tid] = d0 * d0 + d1 * d1;
    __syncthreads();
    #pragma unroll
    for (int off = 128; off > 0; off >>= 1) {
        if (tid < off) red[tid] += red[tid + off];
        __syncthreads();
    }
    const float rstd = rsqrtf(red[0] * (1.0f / H_) + 1e-5f);

    __bf16* orow = out + (size_t)row * H_;
    orow[tid * 2]     = (__bf16)(d0 * rstd * g[tid * 2]     + bta[tid * 2]);
    orow[tid * 2 + 1] = (__bf16)(d1 * rstd * g[tid * 2 + 1] + bta[tid * 2 + 1]);
}

// ---------------------------------------------------------------------------
// Weight fp32 [K,N] -> bf16 transposed [N,K] (WMMA B-fragment friendly)
// ---------------------------------------------------------------------------
__global__ __launch_bounds__(256)
void transpose_w_bf16_kernel(const float* __restrict__ W, __bf16* __restrict__ Wt) {
    const int idx = blockIdx.x * 256 + threadIdx.x;  // over H*H, coalesced read
    const int k = idx / H_, n = idx % H_;
    Wt[(size_t)n * H_ + k] = (__bf16)W[idx];
}

// ---------------------------------------------------------------------------
// WMMA fragment loaders (contiguous k = 16*(lane/16) + s in both A and B, wave32)
// ---------------------------------------------------------------------------
__device__ __forceinline__ v16bf load_a_frag(const __bf16* __restrict__ A,
                                             int m0, int k0, int lane) {
    // A row-major [*, H_]; lane holds row m0+(lane&15)
    const __bf16* p = A + (size_t)(m0 + (lane & 15)) * H_ + k0 + ((lane & 16) ? 16 : 0);
    v8bf lo = *(const v8bf*)p;          // K = sel*16 .. +7
    v8bf hi = *(const v8bf*)(p + 8);    // K = sel*16+8 .. +7
    v16bf r;
    #pragma unroll
    for (int i = 0; i < 8; ++i) { r[i] = lo[i]; r[i + 8] = hi[i]; }
    return r;
}
__device__ __forceinline__ v16bf load_b_frag(const __bf16* __restrict__ Wt,
                                             int n0, int k0, int lane) {
    // Wt is [N,K] row-major: lane (col) reads 16 contiguous K values (32 B)
    const __bf16* p = Wt + (size_t)(n0 + (lane & 15)) * H_ + k0 + ((lane & 16) ? 16 : 0);
    return *(const v16bf*)p;
}

// ---------------------------------------------------------------------------
// Fused gate GEMM: kz = Xn@Wz+bz, hh = Xn@Wh+bh  ->  lc, lv (log-space terms)
// one wave = 64x16 strip of BOTH outputs; B fragments reused across 4 M-tiles
// ---------------------------------------------------------------------------
__global__ __launch_bounds__(256)
void gate_gemm_kernel(const __bf16* __restrict__ Xn,
                      const __bf16* __restrict__ WzT,
                      const __bf16* __restrict__ WhT,
                      const float* __restrict__ bz,
                      const float* __restrict__ bh,
                      float* __restrict__ lc,
                      float* __restrict__ lv) {
    const int lane = threadIdx.x & 31;
    const int wid  = blockIdx.x * (blockDim.x >> 5) + (threadIdx.x >> 5);
    const int m0 = (wid >> 5) * (16 * MBLK);   // 512 M-strips
    const int n0 = (wid & 31) * 16;            // 32 N-tiles

    v8f accz[MBLK] = {}; v8f acch[MBLK] = {};
    #pragma unroll 2
    for (int k0 = 0; k0 < H_; k0 += 32) {
        const v16bf bz_f = load_b_frag(WzT, n0, k0, lane);
        const v16bf bh_f = load_b_frag(WhT, n0, k0, lane);
        #pragma unroll
        for (int i = 0; i < MBLK; ++i) {
            const v16bf a = load_a_frag(Xn, m0 + 16 * i, k0, lane);
            accz[i] = __builtin_amdgcn_wmma_f32_16x16x32_bf16(false, a, false, bz_f,
                                                              (short)0, accz[i], false, false);
            acch[i] = __builtin_amdgcn_wmma_f32_16x16x32_bf16(false, a, false, bh_f,
                                                              (short)0, acch[i], false, false);
        }
    }

    const int col = n0 + (lane & 15);
    const float bzc = bz[col], bhc = bh[col];
    #pragma unroll
    for (int i = 0; i < MBLK; ++i) {
        const int rbase = m0 + 16 * i + ((lane & 16) ? 8 : 0);
        #pragma unroll
        for (int r = 0; r < 8; ++r) {
            const float kz = accz[i][r] + bzc;
            const float hh = acch[i][r] + bhc;
            const float lcv = -softplus_f(kz);    // log(1 - sigmoid(kz))
            const float lzv = -softplus_f(-kz);   // log sigmoid(kz)
            const float lgv = (hh >= 0.0f) ? __logf(hh + 0.5f) : -softplus_f(-hh);
            const size_t addr = (size_t)(rbase + r) * H_ + col;
            lc[addr] = lcv;
            lv[addr] = lzv + lgv;
        }
    }
}

// ---------------------------------------------------------------------------
// Chunked log-space scan, 3 passes. Recurrence lh' = logaddexp(a + lh, v)
// composes as (A,V) o (a,v) = (A+a, logaddexp(a+V, v)).
// ---------------------------------------------------------------------------
__global__ __launch_bounds__(256)
void scan_chunk_reduce_kernel(const float* __restrict__ lc,
                              const float* __restrict__ lv,
                              float* __restrict__ chA,
                              float* __restrict__ chV) {
    const int h  = (blockIdx.x & 1) * 256 + threadIdx.x;
    const int c  = (blockIdx.x >> 1) & (NCHUNK - 1);
    const int b  = blockIdx.x >> 5;                  // /(2*NCHUNK)
    size_t idx = ((size_t)b * T_ + (size_t)c * CHUNK) * H_ + h;
    float A = 0.0f, V = -1e30f;
    for (int t = 0; t < CHUNK; ++t, idx += H_) {
        const float a = lc[idx], v = lv[idx];
        A += a;
        V = logaddexp_f(V + a, v);
    }
    const int ch = b * H_ + h;
    chA[c * BH + ch] = A;
    chV[c * BH + ch] = V;
}

__global__ __launch_bounds__(256)
void scan_chunk_scan_kernel(float* __restrict__ chA, const float* __restrict__ chV) {
    const int ch = blockIdx.x * 256 + threadIdx.x;   // 2048 channels
    float lh = -0.69314718056f;                      // log g(0) = log 0.5
    #pragma unroll
    for (int c = 0; c < NCHUNK; ++c) {
        const float a = chA[c * BH + ch];
        const float v = chV[c * BH + ch];
        chA[c * BH + ch] = lh;                       // incoming state for chunk c
        lh = logaddexp_f(a + lh, v);
    }
}

__global__ __launch_bounds__(256)
void scan_apply_kernel(const float* __restrict__ lc,
                       const float* __restrict__ lv,
                       const float* __restrict__ chIn,
                       const float* __restrict__ x,
                       float* __restrict__ x1) {
    const int h  = (blockIdx.x & 1) * 256 + threadIdx.x;
    const int c  = (blockIdx.x >> 1) & (NCHUNK - 1);
    const int b  = blockIdx.x >> 5;
    const int ch = b * H_ + h;
    size_t idx = ((size_t)b * T_ + (size_t)c * CHUNK) * H_ + h;
    float lh = chIn[c * BH + ch];
    for (int t = 0; t < CHUNK; ++t, idx += H_) {
        lh = logaddexp_f(lc[idx] + lh, lv[idx]);
        x1[idx] = x[idx] + __expf(lh);               // residual: x + minGRU(...)
    }
}

// ---------------------------------------------------------------------------
// FFN GEMM 1: Hrelu = relu(Xn2 @ W1 + b1), stored bf16 for the next WMMA
// 64x16 strip per wave, B fragment reused across M-tiles
// ---------------------------------------------------------------------------
__global__ __launch_bounds__(256)
void ffn1_gemm_kernel(const __bf16* __restrict__ Xn2,
                      const __bf16* __restrict__ W1T,
                      const float* __restrict__ b1,
                      __bf16* __restrict__ Hrelu) {
    const int lane = threadIdx.x & 31;
    const int wid  = blockIdx.x * (blockDim.x >> 5) + (threadIdx.x >> 5);
    const int m0 = (wid >> 5) * (16 * MBLK);
    const int n0 = (wid & 31) * 16;

    v8f acc[MBLK] = {};
    #pragma unroll 2
    for (int k0 = 0; k0 < H_; k0 += 32) {
        const v16bf b = load_b_frag(W1T, n0, k0, lane);
        #pragma unroll
        for (int i = 0; i < MBLK; ++i) {
            const v16bf a = load_a_frag(Xn2, m0 + 16 * i, k0, lane);
            acc[i] = __builtin_amdgcn_wmma_f32_16x16x32_bf16(false, a, false, b,
                                                             (short)0, acc[i], false, false);
        }
    }
    const int col  = n0 + (lane & 15);
    const float bc = b1[col];
    #pragma unroll
    for (int i = 0; i < MBLK; ++i) {
        const int rbase = m0 + 16 * i + ((lane & 16) ? 8 : 0);
        #pragma unroll
        for (int r = 0; r < 8; ++r) {
            Hrelu[(size_t)(rbase + r) * H_ + col] = (__bf16)fmaxf(acc[i][r] + bc, 0.0f);
        }
    }
}

// ---------------------------------------------------------------------------
// FFN GEMM 2 + final residual: out = x1 + Hrelu @ W2 + b2   (fp32 out)
// ---------------------------------------------------------------------------
__global__ __launch_bounds__(256)
void ffn2_gemm_kernel(const __bf16* __restrict__ Hrelu,
                      const __bf16* __restrict__ W2T,
                      const float* __restrict__ b2,
                      const float* __restrict__ x1,
                      float* __restrict__ out) {
    const int lane = threadIdx.x & 31;
    const int wid  = blockIdx.x * (blockDim.x >> 5) + (threadIdx.x >> 5);
    const int m0 = (wid >> 5) * (16 * MBLK);
    const int n0 = (wid & 31) * 16;

    v8f acc[MBLK] = {};
    #pragma unroll 2
    for (int k0 = 0; k0 < H_; k0 += 32) {
        const v16bf b = load_b_frag(W2T, n0, k0, lane);
        #pragma unroll
        for (int i = 0; i < MBLK; ++i) {
            const v16bf a = load_a_frag(Hrelu, m0 + 16 * i, k0, lane);
            acc[i] = __builtin_amdgcn_wmma_f32_16x16x32_bf16(false, a, false, b,
                                                             (short)0, acc[i], false, false);
        }
    }
    const int col  = n0 + (lane & 15);
    const float bc = b2[col];
    #pragma unroll
    for (int i = 0; i < MBLK; ++i) {
        const int rbase = m0 + 16 * i + ((lane & 16) ? 8 : 0);
        #pragma unroll
        for (int r = 0; r < 8; ++r) {
            const size_t addr = (size_t)(rbase + r) * H_ + col;
            out[addr] = x1[addr] + acc[i][r] + bc;
        }
    }
}

// ---------------------------------------------------------------------------
extern "C" void kernel_launch(void* const* d_in, const int* in_sizes, int n_in,
                              void* d_out, int out_size, void* d_ws, size_t ws_size,
                              hipStream_t stream) {
    const float* x   = (const float*)d_in[0];
    const float* g1  = (const float*)d_in[1];
    const float* be1 = (const float*)d_in[2];
    const float* Wz  = (const float*)d_in[3];
    const float* bz  = (const float*)d_in[4];
    const float* Wh  = (const float*)d_in[5];
    const float* bh  = (const float*)d_in[6];
    const float* g2  = (const float*)d_in[7];
    const float* be2 = (const float*)d_in[8];
    const float* W1  = (const float*)d_in[9];
    const float* b1  = (const float*)d_in[10];
    const float* W2  = (const float*)d_in[11];
    const float* b2  = (const float*)d_in[12];
    float* out = (float*)d_out;

    // workspace carve-up (256B aligned)
    char* ws = (char*)d_ws;
    size_t off = 0;
    auto carve = [&](size_t bytes) -> void* {
        void* p = ws + off;
        off += (bytes + 255) & ~(size_t)255;
        return p;
    };
    __bf16* xn    = (__bf16*)carve((size_t)NROW * H_ * 2);   // LN1/LN2 output (reused)
    __bf16* WzT   = (__bf16*)carve((size_t)H_ * H_ * 2);
    __bf16* WhT   = (__bf16*)carve((size_t)H_ * H_ * 2);
    __bf16* W1T   = (__bf16*)carve((size_t)H_ * H_ * 2);
    __bf16* W2T   = (__bf16*)carve((size_t)H_ * H_ * 2);
    float*  lc    = (float*)carve((size_t)NROW * H_ * 4);    // log_coeffs
    float*  lv    = (float*)carve((size_t)NROW * H_ * 4);    // log_z + log_tilde_h
    float*  x1    = (float*)carve((size_t)NROW * H_ * 4);    // after first residual
    __bf16* hrel  = (__bf16*)carve((size_t)NROW * H_ * 2);   // relu FFN hidden
    float*  chA   = (float*)carve((size_t)NCHUNK * BH * 4);
    float*  chV   = (float*)carve((size_t)NCHUNK * BH * 4);

    // 8 waves/block, one 64x16 strip per wave
    const int gemm_blocks = (NROW / (16 * MBLK)) * (H_ / 16) / 8;   // 2048

    // Stage 0: weight prep (bf16, transposed to [N,K])
    transpose_w_bf16_kernel<<<H_ * H_ / 256, 256, 0, stream>>>(Wz, WzT);
    transpose_w_bf16_kernel<<<H_ * H_ / 256, 256, 0, stream>>>(Wh, WhT);
    transpose_w_bf16_kernel<<<H_ * H_ / 256, 256, 0, stream>>>(W1, W1T);
    transpose_w_bf16_kernel<<<H_ * H_ / 256, 256, 0, stream>>>(W2, W2T);

    // Stage 1: LN1 -> fused dual GEMM + gate math
    ln_to_bf16_kernel<<<NROW, 256, 0, stream>>>(x, g1, be1, xn);
    gate_gemm_kernel<<<gemm_blocks, 256, 0, stream>>>(xn, WzT, WhT, bz, bh, lc, lv);

    // Stage 2: chunked log-space scan + residual
    scan_chunk_reduce_kernel<<<B_ * NCHUNK * 2, 256, 0, stream>>>(lc, lv, chA, chV);
    scan_chunk_scan_kernel<<<BH / 256, 256, 0, stream>>>(chA, chV);
    scan_apply_kernel<<<B_ * NCHUNK * 2, 256, 0, stream>>>(lc, lv, chA, x, x1);

    // Stage 3: LN2 -> FFN (GEMM, ReLU, GEMM) + residual
    ln_to_bf16_kernel<<<NROW, 256, 0, stream>>>(x1, g2, be2, xn);
    ffn1_gemm_kernel<<<gemm_blocks, 256, 0, stream>>>(xn, W1T, b1, hrel);
    ffn2_gemm_kernel<<<gemm_blocks, 256, 0, stream>>>(hrel, W2T, b2, x1, out);
}
